// GatedExpertMixture_42872363549116
// MI455X (gfx1250) — matrix operand order repaired and checked by reading, not compiled, not run on hardware
//
#include <hip/hip_runtime.h>
#include <hip/hip_bf16.h>
#include <math.h>

#define E_ 8
#define B_ 4
#define N_ 2048
#define D_ 1024
#define H_ 1024
#define T_ (B_ * N_)   // 8192 tokens

typedef __attribute__((ext_vector_type(16))) _Float16 v16h;
typedef __attribute__((ext_vector_type(8)))  float    v8f;

// ---------------------------------------------------------------------------
// Init: zero output accumulator + expert counters
// ---------------------------------------------------------------------------
__global__ void moe_init_kernel(float* __restrict__ out, size_t n_out,
                                int* __restrict__ cnt) {
    size_t i  = (size_t)blockIdx.x * blockDim.x + threadIdx.x;
    size_t st = (size_t)gridDim.x * blockDim.x;
    for (size_t k = i; k < n_out; k += st) out[k] = 0.0f;
    if (i < E_) cnt[i] = 0;
}

// ---------------------------------------------------------------------------
// Convert x (fp32) -> f16, row-major [T][D]
// ---------------------------------------------------------------------------
__global__ void moe_cvt_x_kernel(const float* __restrict__ x,
                                 _Float16* __restrict__ xh, size_t n) {
    size_t i  = (size_t)blockIdx.x * blockDim.x + threadIdx.x;
    size_t st = (size_t)gridDim.x * blockDim.x;
    for (size_t k = i; k < n; k += st) xh[k] = (_Float16)x[k];
}

// ---------------------------------------------------------------------------
// Transpose+convert one 1024x1024 fp32 matrix per expert to f16:
//   out[e][c][r] = (f16) in[e][r][c]
// Makes the WMMA B-matrix K-dimension contiguous in memory.
// block (32,8), grid (32,32,E)
// ---------------------------------------------------------------------------
__global__ void moe_transpose_cvt_kernel(const float* __restrict__ in,
                                         _Float16* __restrict__ out) {
    __shared__ _Float16 tile[32][33];
    const int e  = blockIdx.z;
    const int r0 = blockIdx.y * 32, c0 = blockIdx.x * 32;
    const float*    ip = in  + (size_t)e * D_ * H_;
    _Float16*       op = out + (size_t)e * D_ * H_;
    const int tx = threadIdx.x, ty = threadIdx.y;
#pragma unroll
    for (int j = 0; j < 32; j += 8)
        tile[ty + j][tx] = (_Float16)ip[(size_t)(r0 + ty + j) * 1024 + (c0 + tx)];
    __syncthreads();
#pragma unroll
    for (int j = 0; j < 32; j += 8)
        op[(size_t)(c0 + ty + j) * 1024 + (r0 + tx)] = tile[tx][ty + j];
}

// ---------------------------------------------------------------------------
// Router: one wave (32 lanes) per token. Exact fp32 logits, top-2 with
// lowest-index tie-break (matches jax.lax.top_k), softmax over the 2 logits.
// Appends (token, weight) to per-expert lists; records per-token slots for
// the deterministic aux-loss reduction.
// ---------------------------------------------------------------------------
__global__ void moe_router_kernel(const float* __restrict__ x,
                                  const float* __restrict__ rw,
                                  const float* __restrict__ rb,
                                  int*   __restrict__ cnt,
                                  int*   __restrict__ tok_list,
                                  float* __restrict__ w_list,
                                  int*   __restrict__ eslot,
                                  float* __restrict__ wslot) {
    const int wave  = threadIdx.x >> 5;
    const int lane  = threadIdx.x & 31;
    const int token = blockIdx.x * 8 + wave;
    if (token >= T_) return;

    const float* xr = x + (size_t)token * D_;
    float s[E_];
#pragma unroll
    for (int e = 0; e < E_; e++) s[e] = 0.0f;

    for (int d = lane; d < D_; d += 32) {
        const float xv = xr[d];
#pragma unroll
        for (int e = 0; e < E_; e++) s[e] += xv * rw[e * D_ + d];
    }
#pragma unroll
    for (int e = 0; e < E_; e++)
        for (int off = 16; off > 0; off >>= 1)
            s[e] += __shfl_xor(s[e], off, 32);

    if (lane == 0) {
#pragma unroll
        for (int e = 0; e < E_; e++) s[e] += rb[e];
        int i0 = 0;
        for (int e = 1; e < E_; e++) if (s[e] > s[i0]) i0 = e;
        int i1 = -1;
        for (int e = 0; e < E_; e++) {
            if (e == i0) continue;
            if (i1 < 0 || s[e] > s[i1]) i1 = e;
        }
        const float l0 = s[i0], l1 = s[i1];
        const float e1 = expf(l1 - l0);          // l0 >= l1
        const float inv = 1.0f / (1.0f + e1);
        const float w0 = inv, w1 = e1 * inv;

        int p0 = atomicAdd(&cnt[i0], 1);
        tok_list[i0 * T_ + p0] = token; w_list[i0 * T_ + p0] = w0;
        int p1 = atomicAdd(&cnt[i1], 1);
        tok_list[i1 * T_ + p1] = token; w_list[i1 * T_ + p1] = w1;

        eslot[token * 2 + 0] = i0; wslot[token * 2 + 0] = w0;
        eslot[token * 2 + 1] = i1; wslot[token * 2 + 1] = w1;
    }
}

// ---------------------------------------------------------------------------
// Aux loss: aux = E * sum_e (cnt_e/T) * (sum_t w_{t,e} / T)
//         = E/T^2 * sum over (token,slot) of cnt[e_slot] * w_slot
// Single block, fixed-order tree reduction => deterministic.
// ---------------------------------------------------------------------------
__global__ void moe_aux_kernel(const int* __restrict__ cnt,
                               const int* __restrict__ eslot,
                               const float* __restrict__ wslot,
                               float* __restrict__ out_aux) {
    __shared__ float red[256];
    float a = 0.0f;
    for (int i = threadIdx.x; i < T_ * 2; i += 256)
        a += (float)cnt[eslot[i]] * wslot[i];
    red[threadIdx.x] = a;
    __syncthreads();
    for (int sft = 128; sft > 0; sft >>= 1) {
        if ((int)threadIdx.x < sft) red[threadIdx.x] += red[threadIdx.x + sft];
        __syncthreads();
    }
    if (threadIdx.x == 0)
        out_aux[0] = (float)E_ * red[0] / ((float)T_ * (float)T_);
}

// ---------------------------------------------------------------------------
// Expert FFN: block = (tile of 16 tokens, expert). 256 threads = 8 waves.
// Wave w owns output columns [w*128, w*128+128) as 8 WMMA 16x16 tiles.
//   up:   h = gelu(x @ W_up + b_up)   (K = D = 1024, 32 k-steps of 32)
//   down: y = h @ W_down + b_down     (K = H = 1024)
// A fragments from LDS (two ds_read_b128 per ISA 16-bit A layout),
// B fragments from global f16 K-contiguous weights (two global b128).
// Scatter: out[token] += router_w * y  via global_atomic_add_f32.
// ---------------------------------------------------------------------------
__global__ __launch_bounds__(256)
void moe_ffn_kernel(const _Float16* __restrict__ xh,
                    const _Float16* __restrict__ Wup_t,   // [E][H][D]
                    const _Float16* __restrict__ Wdn_t,   // [E][D][H]
                    const float* __restrict__ b_up,       // [E][H]
                    const float* __restrict__ b_dn,       // [E][D]
                    const int*   __restrict__ cnt,
                    const int*   __restrict__ tok_list,
                    const float* __restrict__ w_list,
                    float* __restrict__ out) {
    __shared__ _Float16 Sbuf[16 * 1024];   // x tile, then reused for gelu(h)
    __shared__ int   tokL[16];
    __shared__ float wgtL[16];

    const int e    = blockIdx.y;
    const int tile = blockIdx.x;
    const int c    = cnt[e];
    if (tile * 16 >= c) return;
    const int rows = min(16, c - tile * 16);
    const int tid  = threadIdx.x;

    if (tid < 16) {
        if (tid < rows) {
            tokL[tid] = tok_list[e * T_ + tile * 16 + tid];
            wgtL[tid] = w_list[e * T_ + tile * 16 + tid];
        } else {
            tokL[tid] = -1; wgtL[tid] = 0.0f;
        }
    }
    __syncthreads();

    // Stage A = xh rows of this tile: 16 rows x 1024 halfs, 16B chunks
    for (int ch = tid; ch < 16 * 128; ch += 256) {
        const int m   = ch >> 7;
        const int off = (ch & 127) * 8;         // in halfs
        uint4 v = make_uint4(0u, 0u, 0u, 0u);
        const int t = tokL[m];
        if (t >= 0) v = *(const uint4*)(xh + (size_t)t * D_ + off);
        *(uint4*)(&Sbuf[m * 1024 + off]) = v;
    }
    __syncthreads();

    const int wave  = tid >> 5;
    const int lane  = tid & 31;
    const int hi    = lane >> 4;     // which 16-lane half
    const int nl    = lane & 15;
    const int wbase = wave * 128;

    union Frag { uint4 u[2]; v16h v; };
    v8f acc[8];
    v8f zero = {};
#pragma unroll
    for (int nt = 0; nt < 8; nt++) acc[nt] = zero;

    // -------- up projection: K over D --------
    const _Float16* wup_e = Wup_t + (size_t)e * D_ * H_;
    for (int k0 = 0; k0 < D_; k0 += 32) {
        Frag a;
        const _Float16* ap = &Sbuf[nl * 1024 + k0 + hi * 8];
        a.u[0] = *(const uint4*)(ap);
        a.u[1] = *(const uint4*)(ap + 16);
#pragma unroll
        for (int nt = 0; nt < 8; nt++) {
            const int n = wbase + nt * 16 + nl;
            const _Float16* bp = wup_e + (size_t)n * D_ + k0 + hi * 8;
            Frag b;
            b.u[0] = *(const uint4*)(bp);
            b.u[1] = *(const uint4*)(bp + 16);
            acc[nt] = __builtin_amdgcn_wmma_f32_16x16x32_f16(
                false, a.v, false, b.v, (short)0, acc[nt], false, false);
        }
    }
    __syncthreads();   // everyone done reading x tile

    // bias + exact-erf gelu, write f16 h into Sbuf
#pragma unroll
    for (int nt = 0; nt < 8; nt++) {
        const int n  = wbase + nt * 16 + nl;
        const float bu = b_up[e * H_ + n];
#pragma unroll
        for (int i = 0; i < 8; i++) {
            const int m = i + hi * 8;
            const float v = acc[nt][i] + bu;
            const float g = 0.5f * v * (1.0f + erff(v * 0.70710678118654752440f));
            Sbuf[m * 1024 + n] = (_Float16)g;
        }
    }
    __syncthreads();

    // -------- down projection: K over H --------
#pragma unroll
    for (int nt = 0; nt < 8; nt++) acc[nt] = zero;
    const _Float16* wdn_e = Wdn_t + (size_t)e * D_ * H_;
    for (int k0 = 0; k0 < H_; k0 += 32) {
        Frag a;
        const _Float16* ap = &Sbuf[nl * 1024 + k0 + hi * 8];
        a.u[0] = *(const uint4*)(ap);
        a.u[1] = *(const uint4*)(ap + 16);
#pragma unroll
        for (int nt = 0; nt < 8; nt++) {
            const int n = wbase + nt * 16 + nl;
            const _Float16* bp = wdn_e + (size_t)n * H_ + k0 + hi * 8;
            Frag b;
            b.u[0] = *(const uint4*)(bp);
            b.u[1] = *(const uint4*)(bp + 16);
            acc[nt] = __builtin_amdgcn_wmma_f32_16x16x32_f16(
                false, a.v, false, b.v, (short)0, acc[nt], false, false);
        }
    }

    // epilogue: weighted scatter-add (exactly 2 commutative adds per element)
#pragma unroll
    for (int nt = 0; nt < 8; nt++) {
        const int n  = wbase + nt * 16 + nl;
        const float bd = b_dn[e * D_ + n];
#pragma unroll
        for (int i = 0; i < 8; i++) {
            const int m = i + hi * 8;
            if (m < rows) {
                const float val = wgtL[m] * (acc[nt][i] + bd);
                atomicAdd(&out[(size_t)tokL[m] * D_ + n], val);
            }
        }
    }
}

// ---------------------------------------------------------------------------
extern "C" void kernel_launch(void* const* d_in, const int* in_sizes, int n_in,
                              void* d_out, int out_size, void* d_ws, size_t ws_size,
                              hipStream_t stream) {
    const float* x        = (const float*)d_in[0];
    const float* router_w = (const float*)d_in[1];
    const float* router_b = (const float*)d_in[2];
    const float* W_up     = (const float*)d_in[3];
    const float* b_up     = (const float*)d_in[4];
    const float* W_down   = (const float*)d_in[5];
    const float* b_down   = (const float*)d_in[6];
    float* out = (float*)d_out;

    // workspace carve-up (~50.5 MB)
    char* ws = (char*)d_ws;
    size_t off = 0;
    _Float16* Wup_t = (_Float16*)(ws + off); off += (size_t)E_ * D_ * H_ * 2;
    _Float16* Wdn_t = (_Float16*)(ws + off); off += (size_t)E_ * D_ * H_ * 2;
    _Float16* xh    = (_Float16*)(ws + off); off += (size_t)T_ * D_ * 2;
    int*   cnt      = (int*)  (ws + off);    off += 256;
    int*   tok_list = (int*)  (ws + off);    off += (size_t)E_ * T_ * 4;
    float* w_list   = (float*)(ws + off);    off += (size_t)E_ * T_ * 4;
    int*   eslot    = (int*)  (ws + off);    off += (size_t)T_ * 2 * 4;
    float* wslot    = (float*)(ws + off);    off += (size_t)T_ * 2 * 4;

    const size_t n_out_main = (size_t)T_ * D_;

    moe_init_kernel<<<1024, 256, 0, stream>>>(out, n_out_main, cnt);
    moe_cvt_x_kernel<<<4096, 256, 0, stream>>>(x, xh, (size_t)T_ * D_);

    dim3 tb(32, 8), tg(32, 32, E_);
    moe_transpose_cvt_kernel<<<tg, tb, 0, stream>>>(W_up,   Wup_t);
    moe_transpose_cvt_kernel<<<tg, tb, 0, stream>>>(W_down, Wdn_t);

    moe_router_kernel<<<T_ / 8, 256, 0, stream>>>(x, router_w, router_b,
                                                  cnt, tok_list, w_list,
                                                  eslot, wslot);
    moe_aux_kernel<<<1, 256, 0, stream>>>(cnt, eslot, wslot, out + n_out_main);

    dim3 fg(T_ / 16, E_);   // per expert: at most ceil(T/16) tiles; extra exit early
    moe_ffn_kernel<<<fg, 256, 0, stream>>>(xh, Wup_t, Wdn_t, b_up, b_down,
                                           cnt, tok_list, w_list, out);
}
